// ADAG_9345848836316
// MI455X (gfx1250) — matrix operand order repaired and checked by reading, not compiled, not run on hardware
//
#include <hip/hip_runtime.h>
#include <hip/hip_bf16.h>

typedef __attribute__((ext_vector_type(16))) _Float16 v16h;
typedef __attribute__((ext_vector_type(8)))  _Float16 v8h;
typedef __attribute__((ext_vector_type(8)))  float    v8f;

#define WMMA_F16(A,B,C) __builtin_amdgcn_wmma_f32_16x16x32_f16(false,(A),false,(B),(short)0,(C),false,false)

#define NB 2048      // batch
#define SS 32        // nodes per batch
#define FF 1433      // feature dim
#define HH 64        // hidden
#define KFULL 44     // full 32-wide K chunks of 1433
#define KCH 45       // total chunks (incl. tail)
#define LROW 72      // padded LDS row stride (f16) to avoid bank conflicts

// ---------------------------------------------------------------------------
// Prep: swizzle weights to f16 WMMA B-layout, compute score vectors u1/u2/u3
// B-layout element (chunk c, ntile nt, lane l, i):
//   k = c*32 + (l<16 ? 0 : 8) + (i&7) + (i>>3)*16 ;  n = nt*16 + (l%16)
// stored at ((c*4+nt)*32 + l)*16 + i  -> one 32B contiguous load per lane.
// ---------------------------------------------------------------------------
__global__ __launch_bounds__(256) void k_prep(
    const float* __restrict__ W1, const float* __restrict__ W2,
    const float* __restrict__ Wg1, const float* __restrict__ Wg2,
    const float* __restrict__ vnode, const float* __restrict__ vnode1,
    const float* __restrict__ snoise, const float* __restrict__ snoise1,
    const float* __restrict__ Wn, const float* __restrict__ bn,
    const float* __restrict__ Wn1, const float* __restrict__ bn1,
    const float* __restrict__ Wb1, const float* __restrict__ Wb2,
    const float* __restrict__ Wb3,
    const float* __restrict__ a1p, const float* __restrict__ a2p,
    _Float16* __restrict__ B1, _Float16* __restrict__ B2,
    _Float16* __restrict__ BG1, _Float16* __restrict__ BG2,
    float* __restrict__ U)
{
    const int tid = threadIdx.x;
    // --- W_feat1 (64 x 1433) -> B1, 45 chunks
    for (int idx = tid; idx < KCH*4*32*16; idx += 256) {
        int i  = idx & 15;
        int l  = (idx >> 4) & 31;
        int nt = (idx >> 9) & 3;
        int c  = idx >> 11;
        int k  = c*32 + ((l < 16) ? 0 : 8) + (i & 7) + ((i >> 3) << 4);
        int n  = nt*16 + (l & 15);
        B1[idx] = (_Float16)((k < FF) ? W1[n*FF + k] : 0.0f);
    }
    // --- 64x64 weights -> 2 chunks each
    for (int idx = tid; idx < 2*4*32*16; idx += 256) {
        int i  = idx & 15;
        int l  = (idx >> 4) & 31;
        int nt = (idx >> 9) & 3;
        int c  = idx >> 11;
        int k  = c*32 + ((l < 16) ? 0 : 8) + (i & 7) + ((i >> 3) << 4);
        int n  = nt*16 + (l & 15);
        B2[idx]  = (_Float16)W2 [n*HH + k];
        BG1[idx] = (_Float16)Wg1[n*HH + k];
        BG2[idx] = (_Float16)Wg2[n*HH + k];
    }
    // --- score vectors
    __shared__ float vn1s[HH], vns[HH], pns[HH], rns[HH], u1s[HH], u2s[HH];
    if (tid < HH) {
        float d1 = 0.f, d2 = 0.f, d3 = 0.f, d4 = 0.f;
        for (int j = 0; j < HH; ++j) {
            d1 += vnode1[j] * Wg1[tid*HH + j];
            d2 += vnode [j] * Wg2[tid*HH + j];
            d3 += snoise1[j] * Wn [tid*HH + j];
            d4 += snoise [j] * Wn1[tid*HH + j];
        }
        float a1 = *a1p, a2 = *a2p;
        vn1s[tid] = (d1 >= 0.f) ? d1 : a1 * d1;
        vns [tid] = (d2 >= 0.f) ? d2 : a2 * d2;
        pns [tid] = d3 + bn [tid];
        rns [tid] = d4 + bn1[tid];
    }
    __syncthreads();
    if (tid < HH) {
        float s1 = 0.f, s2 = 0.f, s3 = 0.f;
        for (int i = 0; i < HH; ++i) {
            s1 += vn1s[i] * Wb1[i*HH + tid];
            s2 += vns [i] * Wb2[i*HH + tid];
            s3 += vns [i] * Wb3[i*HH + tid];
        }
        U[tid] = s1; U[HH + tid] = s2; U[2*HH + tid] = s3;
        u1s[tid] = s1; u2s[tid] = s2;
    }
    __syncthreads();
    if (tid == 0) {
        float o1 = 0.f, o2 = 0.f;
        for (int j = 0; j < HH; ++j) { o1 += u1s[j]*pns[j]; o2 += u2s[j]*rns[j]; }
        U[3*HH] = o1;      // noise-pool score offset
        U[3*HH + 1] = o2;  // noise-root score offset
    }
}

// 64-wide f16 WMMA layer from this wave's LDS tile (rows 0..31, stride LROW)
__device__ __forceinline__ void lds_layer64(const _Float16* H,
                                            const _Float16* __restrict__ Bw,
                                            int lane, int mrow, int klo,
                                            v8f acc[2][4])
{
    #pragma unroll
    for (int kc = 0; kc < 2; ++kc) {
        v16h a[2];
        #pragma unroll
        for (int mt = 0; mt < 2; ++mt) {
            const _Float16* rp = H + (mt*16 + mrow)*LROW + kc*32 + klo;
            v8h g0 = *(const v8h*)(rp);
            v8h g1 = *(const v8h*)(rp + 16);
            #pragma unroll
            for (int j = 0; j < 8; ++j) { a[mt][j] = g0[j]; a[mt][8+j] = g1[j]; }
        }
        #pragma unroll
        for (int nt = 0; nt < 4; ++nt) {
            v16h bb = *(const v16h*)(Bw + (((kc*4 + nt)*32 + lane) << 4));
            acc[0][nt] = WMMA_F16(a[0], bb, acc[0][nt]);
            acc[1][nt] = WMMA_F16(a[1], bb, acc[1][nt]);
        }
    }
}

// ---------------------------------------------------------------------------
// Fused gather + feat(layer1,layer2) + W_g1/prelu + pooling + x0 extraction.
// One wave == one batch (32 rows = 2 M-tiles).  2112 waves = 264 blocks.
// Blocks [0,256): node batches.  Blocks [256,264): malicious rows (feat only).
// ---------------------------------------------------------------------------
__global__ __launch_bounds__(256) void k_fused(
    const float* __restrict__ emb,
    const int* __restrict__ node_ids,
    const int* __restrict__ mal_ids,
    const _Float16* __restrict__ B1,
    const _Float16* __restrict__ B2,
    const _Float16* __restrict__ BG1,
    const float* __restrict__ bias1v,
    const float* __restrict__ bias2v,
    const float* __restrict__ a1p,
    float* __restrict__ out_pooled,   // d_out + 5*NB, row-major (NB x 64)
    _Float16* __restrict__ x0out,     // ws: x[:,0,:] as f16 (NB x 64)
    float* __restrict__ malout)       // ws: feat(mal rows) f32 (NB x 64)
{
    __shared__ __align__(16) _Float16 ldsH[8][SS * LROW];
    const int lane  = threadIdx.x & 31;
    const int wv    = threadIdx.x >> 5;
    const int gw    = blockIdx.x * 8 + wv;      // 0..2111
    const bool isMal = (gw >= NB);
    const int mrow  = lane & 15;
    const int klo   = (lane < 16) ? 0 : 8;      // A-matrix K sub-offset
    const int mhalf = (lane < 16) ? 0 : 8;      // C-matrix M offset

    int rid0, rid1;
    if (!isMal) {
        rid0 = node_ids[gw*SS + mrow];
        rid1 = node_ids[gw*SS + 16 + mrow];
    } else {
        int w = gw - NB;
        rid0 = mal_ids[w*SS + mrow];
        rid1 = mal_ids[w*SS + 16 + mrow];
    }
    const float* rp0 = emb + (size_t)rid0 * FF;
    const float* rp1 = emb + (size_t)rid1 * FF;

    // ---- layer 1: K = 1433 ----
    v8f acc[2][4] = {};
    for (int c = 0; c < KFULL; ++c) {
        const int kb = c*32 + klo;
        v16h a0, a1;
        #pragma unroll
        for (int g = 0; g < 2; ++g)
        #pragma unroll
        for (int j = 0; j < 8; ++j) {
            int k = kb + g*16 + j;
            a0[g*8+j] = (_Float16)rp0[k];
            a1[g*8+j] = (_Float16)rp1[k];
        }
        #pragma unroll
        for (int nt = 0; nt < 4; ++nt) {
            v16h bb = *(const v16h*)(B1 + (((c*4 + nt)*32 + lane) << 4));
            acc[0][nt] = WMMA_F16(a0, bb, acc[0][nt]);
            acc[1][nt] = WMMA_F16(a1, bb, acc[1][nt]);
        }
    }
    {   // tail chunk c = 44 (k in [1408,1440), valid < 1433)
        const int kb = KFULL*32 + klo;
        v16h a0, a1;
        #pragma unroll
        for (int g = 0; g < 2; ++g)
        #pragma unroll
        for (int j = 0; j < 8; ++j) {
            int k = kb + g*16 + j;
            a0[g*8+j] = (k < FF) ? (_Float16)rp0[k] : (_Float16)0.0f;
            a1[g*8+j] = (k < FF) ? (_Float16)rp1[k] : (_Float16)0.0f;
        }
        #pragma unroll
        for (int nt = 0; nt < 4; ++nt) {
            v16h bb = *(const v16h*)(B1 + (((KFULL*4 + nt)*32 + lane) << 4));
            acc[0][nt] = WMMA_F16(a0, bb, acc[0][nt]);
            acc[1][nt] = WMMA_F16(a1, bb, acc[1][nt]);
        }
    }

    // h1 = relu(acc + b1) -> LDS (f16, C-layout scatter to row-major)
    _Float16* H = &ldsH[wv][0];
    #pragma unroll
    for (int nt = 0; nt < 4; ++nt) {
        const float bv = bias1v[nt*16 + mrow];
        #pragma unroll
        for (int mt = 0; mt < 2; ++mt)
        #pragma unroll
        for (int r = 0; r < 8; ++r) {
            float v = acc[mt][nt][r] + bv;
            v = (v > 0.0f) ? v : 0.0f;
            H[(mt*16 + r + mhalf)*LROW + nt*16 + mrow] = (_Float16)v;
        }
    }
    __syncthreads();

    // ---- layer 2: h2 = h1 @ W_feat2.T + b2 ----
    v8f acc2[2][4] = {};
    lds_layer64(H, B2, lane, mrow, klo, acc2);

    if (!isMal) {
        __syncthreads();
        #pragma unroll
        for (int nt = 0; nt < 4; ++nt) {
            const float bv = bias2v[nt*16 + mrow];
            #pragma unroll
            for (int mt = 0; mt < 2; ++mt)
            #pragma unroll
            for (int r = 0; r < 8; ++r) {
                float v = acc2[mt][nt][r] + bv;
                H[(mt*16 + r + mhalf)*LROW + nt*16 + mrow] = (_Float16)v;
            }
        }
        __syncthreads();

        // ---- x = prelu(h2 @ W_g1.T, a1) ----
        v8f acc3[2][4] = {};
        lds_layer64(H, BG1, lane, mrow, klo, acc3);
        const float a1s = *a1p;

        // pooling over s=1..31 (wave-local) + extract row s=0
        #pragma unroll
        for (int nt = 0; nt < 4; ++nt) {
            float colsum = 0.0f, x00 = 0.0f;
            #pragma unroll
            for (int mt = 0; mt < 2; ++mt)
            #pragma unroll
            for (int r = 0; r < 8; ++r) {
                float v = acc3[mt][nt][r];
                v = (v >= 0.0f) ? v : a1s * v;
                if (mt == 0 && r == 0 && lane < 16) x00 = v;   // s==0 row
                else colsum += v;
            }
            colsum += __shfl_xor(colsum, 16);
            if (lane < 16) {
                out_pooled[gw*HH + nt*16 + lane] = colsum * (1.0f/31.0f);
                x0out[gw*HH + nt*16 + lane] = (_Float16)x00;
            }
        }
    } else {
        const int w = gw - NB;
        #pragma unroll
        for (int nt = 0; nt < 4; ++nt) {
            const float bv = bias2v[nt*16 + mrow];
            #pragma unroll
            for (int mt = 0; mt < 2; ++mt)
            #pragma unroll
            for (int r = 0; r < 8; ++r) {
                int row = w*SS + mt*16 + r + mhalf;
                malout[row*HH + nt*16 + mrow] = acc2[mt][nt][r] + bv;
            }
        }
    }
}

// ---------------------------------------------------------------------------
// root = prelu(x0 @ W_g2.T, a2), 16 batches per wave, 128 waves.
// ---------------------------------------------------------------------------
__global__ __launch_bounds__(256) void k_root(
    const _Float16* __restrict__ x0in,
    const _Float16* __restrict__ BG2,
    const float* __restrict__ a2p,
    float* __restrict__ rootout)
{
    const int lane  = threadIdx.x & 31;
    const int wv    = threadIdx.x >> 5;
    const int b0    = (blockIdx.x * 8 + wv) * 16;
    const int mrow  = lane & 15;
    const int klo   = (lane < 16) ? 0 : 8;
    const int mhalf = klo;
    v8f acc[4] = {};
    #pragma unroll
    for (int kc = 0; kc < 2; ++kc) {
        const _Float16* rp = x0in + (b0 + mrow)*HH + kc*32 + klo;
        v8h g0 = *(const v8h*)(rp);
        v8h g1 = *(const v8h*)(rp + 16);
        v16h a;
        #pragma unroll
        for (int j = 0; j < 8; ++j) { a[j] = g0[j]; a[8+j] = g1[j]; }
        #pragma unroll
        for (int nt = 0; nt < 4; ++nt) {
            v16h bb = *(const v16h*)(BG2 + (((kc*4 + nt)*32 + lane) << 4));
            acc[nt] = WMMA_F16(a, bb, acc[nt]);
        }
    }
    const float a2s = *a2p;
    #pragma unroll
    for (int nt = 0; nt < 4; ++nt)
    #pragma unroll
    for (int r = 0; r < 8; ++r) {
        float v = acc[nt][r];
        v = (v >= 0.0f) ? v : a2s * v;
        rootout[(b0 + r + mhalf)*HH + nt*16 + mrow] = v;
    }
}

// ---------------------------------------------------------------------------
// Five score arrays; noise variants are constant offsets (linearity).
// ---------------------------------------------------------------------------
__global__ __launch_bounds__(256) void k_scores(
    const float* __restrict__ U,         // u1[64], u2[64], u3[64], off1, off2
    const float* __restrict__ pooled,
    const float* __restrict__ root,
    const float* __restrict__ mal,
    const float* __restrict__ bb1, const float* __restrict__ bb2,
    const float* __restrict__ bb3,
    float* __restrict__ out)
{
    const int b = blockIdx.x * 256 + threadIdx.x;
    if (b >= NB) return;
    float s1 = 0.f, s2 = 0.f, s3r = 0.f, s3m = 0.f;
    for (int j = 0; j < HH; ++j) {
        float p = pooled[b*HH + j];
        float r = root  [b*HH + j];
        float m = mal   [b*HH + j];
        s1  += U[j]        * p;
        s2  += U[HH + j]   * r;
        s3r += U[2*HH + j] * r;
        s3m += U[2*HH + j] * m;
    }
    out[b]        = s1 + bb1[0];                       // pool_score
    out[NB + b]   = s1 + bb1[0] + U[3*HH];             // noise_pool_score
    out[2*NB + b] = s2 + bb2[0];                       // root_score
    out[3*NB + b] = s2 + bb2[0] + U[3*HH + 1];         // noise_root_score
    out[4*NB + b] = 0.5f*(s3r + s3m) + bb3[0];         // malicious_score
}

extern "C" void kernel_launch(void* const* d_in, const int* in_sizes, int n_in,
                              void* d_out, int out_size, void* d_ws, size_t ws_size,
                              hipStream_t stream) {
    const float* emb     = (const float*)d_in[0];
    const int*   node_ids= (const int*)  d_in[1];
    const int*   mal_ids = (const int*)  d_in[2];
    const float* W1      = (const float*)d_in[3];
    const float* b1      = (const float*)d_in[4];
    const float* W2      = (const float*)d_in[5];
    const float* b2      = (const float*)d_in[6];
    const float* Wg1     = (const float*)d_in[7];
    const float* a1      = (const float*)d_in[8];
    const float* Wg2     = (const float*)d_in[9];
    const float* a2      = (const float*)d_in[10];
    const float* vnode   = (const float*)d_in[11];
    const float* vnode1  = (const float*)d_in[12];
    const float* snoise  = (const float*)d_in[13];
    const float* snoise1 = (const float*)d_in[14];
    const float* Wn      = (const float*)d_in[15];
    const float* bn      = (const float*)d_in[16];
    const float* Wn1     = (const float*)d_in[17];
    const float* bn1     = (const float*)d_in[18];
    const float* Wb1     = (const float*)d_in[19];
    const float* bb1     = (const float*)d_in[20];
    const float* Wb2     = (const float*)d_in[21];
    const float* bb2     = (const float*)d_in[22];
    const float* Wb3     = (const float*)d_in[23];
    const float* bb3     = (const float*)d_in[24];

    char* ws = (char*)d_ws;
    _Float16* B1  = (_Float16*)(ws + 0);        // 45*4*512 f16 = 184320 B
    _Float16* B2h = (_Float16*)(ws + 184320);   // 8192 B
    _Float16* BG1 = (_Float16*)(ws + 192512);   // 8192 B
    _Float16* BG2 = (_Float16*)(ws + 200704);   // 8192 B
    _Float16* X0  = (_Float16*)(ws + 208896);   // 2048*64 f16 = 262144 B
    float*    MAL = (float*)   (ws + 471040);   // 2048*64 f32 = 524288 B
    float*    ROOT= (float*)   (ws + 995328);   // 2048*64 f32 = 524288 B
    float*    U   = (float*)   (ws + 1519616);  // 194 f32

    float* out = (float*)d_out;
    float* out_pooled = out + 5*NB;             // pooled is written in-place

    hipLaunchKernelGGL(k_prep, dim3(1), dim3(256), 0, stream,
        W1, W2, Wg1, Wg2, vnode, vnode1, snoise, snoise1,
        Wn, bn, Wn1, bn1, Wb1, Wb2, Wb3, a1, a2, B1, B2h, BG1, BG2, U);

    hipLaunchKernelGGL(k_fused, dim3(264), dim3(256), 0, stream,
        emb, node_ids, mal_ids, B1, B2h, BG1, b1, b2, a1,
        out_pooled, X0, MAL);

    hipLaunchKernelGGL(k_root, dim3(16), dim3(256), 0, stream, X0, BG2, a2, ROOT);

    hipLaunchKernelGGL(k_scores, dim3(8), dim3(256), 0, stream,
        U, out_pooled, ROOT, MAL, bb1, bb2, bb3, out);
}
